// Non_local_block_71691594105157
// MI455X (gfx1250) — matrix operand order repaired and checked by reading, not compiled, not run on hardware
//
#include <hip/hip_runtime.h>
#include <hip/hip_bf16.h>

// MI455X / gfx1250, wave32. All GEMMs on v_wmma_f32_16x16x32_bf16.
// - Attention streams key/value chunks HBM->LDS with global_load_async_to_lds_b128
//   (ASYNCcnt, double-buffered), overlapping DMA with WMMA.
// - sched_group_barrier pipelines DS-read groups against WMMA groups so each
//   WMMA run retires on a partial s_wait_dscnt instead of a full drain.

typedef __attribute__((ext_vector_type(16))) __bf16 v16bf;
typedef __attribute__((ext_vector_type(8)))  float  v8f;

#define NPIX 3136                       // 56*56
#define SECN ((size_t)8 * NPIX * 128)   // one (B, N, 128) bf16 plane, in ushorts

#if defined(__has_builtin)
#if __has_builtin(__builtin_amdgcn_sched_group_barrier)
#define SGB(mask, n) __builtin_amdgcn_sched_group_barrier(mask, n, 0)
#endif
#endif
#ifndef SGB
#define SGB(mask, n)
#endif
#define SGB_DSR 0x100   // DS read
#define SGB_DSW 0x200   // DS write
#define SGB_MMA 0x008   // MFMA/WMMA

__device__ __forceinline__ unsigned short f2bf(float x) {
  unsigned int b = __float_as_uint(x);
  b += 0x7FFFu + ((b >> 16) & 1u);      // round-to-nearest-even
  return (unsigned short)(b >> 16);
}

union FragU { unsigned int u32[8]; v16bf v; };

// A fragment (16x32 bf16) from row-major [M][K] LDS; ISA 7.12.2:
// lane m=lane&15, h=lane>>4; VGPR j: K = k0 + (j&3)*2 + (j>>2)*16 + 8h (pair).
__device__ __forceinline__ v16bf load_frag_a(const unsigned short* base, int row_len,
                                             int m_base, int k0, int lane) {
  const int m = m_base + (lane & 15);
  const int h = lane >> 4;
  const unsigned short* p = base + m * row_len + k0 + 8 * h;
  FragU f;
#pragma unroll
  for (int j = 0; j < 8; ++j) {
    int k = ((j & 3) << 1) + ((j >> 2) << 4);
    f.u32[j] = *(const unsigned int*)(p + k);
  }
  return f.v;
}

// B fragment (32x16 bf16) from [col][K] LDS; VGPR j: K = k0 + (j&3)*2 + (j>>2)*16 + 8h (pair).
__device__ __forceinline__ v16bf load_frag_b(const unsigned short* base, int row_len,
                                             int col_base, int k0, int lane) {
  const int c = col_base + (lane & 15);
  const int h = lane >> 4;
  const unsigned short* p = base + c * row_len + k0 + 8 * h;
  FragU f;
#pragma unroll
  for (int j = 0; j < 8; ++j) {
    f.u32[j] = *(const unsigned int*)(p + ((j & 3) << 1) + ((j >> 2) << 4));
  }
  return f.v;
}

__device__ __forceinline__ v8f wmma_bf16(v16bf a, v16bf b, v8f c) {
  return __builtin_amdgcn_wmma_f32_16x16x32_bf16(false, a, false, b, (short)0, c, false, false);
}

// CDNA5 async HBM->LDS: 16 bytes/lane, tracked by ASYNCcnt (not LOADcnt).
// saddr form: both LDS and global take VGPR byte offsets (same IOFFSET=0).
__device__ __forceinline__ void async_copy16(unsigned lds_addr, unsigned voff,
                                             const void* sbase) {
  asm volatile("global_load_async_to_lds_b128 %0, %1, %2"
               :: "v"(lds_addr), "v"(voff), "s"(sbase) : "memory");
}
__device__ __forceinline__ void wait_async_le16() {
  asm volatile("s_wait_asynccnt 16" ::: "memory");
}
__device__ __forceinline__ void wait_async_0() {
  asm volatile("s_wait_asynccnt 0" ::: "memory");
}

// Pipelined DS-read / WMMA pattern for a region of NK k-steps with
// LDG DS reads and NMMA WMMAs per k-step: loads of k+1 overlap WMMAs of k.
// Template parameters so the builtin sees integer constant expressions.
template <int NK, int LDG, int NMMA>
__device__ __forceinline__ void sgb_pipeline() {
  SGB(SGB_DSR, LDG);
#pragma unroll
  for (int k = 0; k < NK - 1; ++k) {
    SGB(SGB_DSR, LDG);
    SGB(SGB_MMA, NMMA);
  }
  SGB(SGB_MMA, NMMA);
}

// ---------------------------------------------------------------------------
// Kernel 1: fused theta/phi/g projections (stacked 384x256 weight).
// theta, phi stored (b, n, ch) bf16; g stored (b, ch, n) bf16 so the attention
// kernel's [ch][m] tile is a linear per-row async copy.
// ---------------------------------------------------------------------------
__global__ __launch_bounds__(128) void proj_kernel(
    const float* __restrict__ x, const float* __restrict__ w_theta,
    const float* __restrict__ w_phi, const float* __restrict__ w_g,
    unsigned short* __restrict__ tpg) {
  __shared__ __align__(16) unsigned short w_lds[64 * 128];    // 16 KB, K half
  __shared__ __align__(16) unsigned short x_lds[112 * 128];   // 28 KB, [n][c] half
  const int tid = threadIdx.x, lane = tid & 31, wave = tid >> 5;
  const int n0 = blockIdx.x * 112;
  const int o0 = blockIdx.y * 64;
  const int b  = blockIdx.z;

  v8f acc[7];
  v8f zero = {};
#pragma unroll
  for (int t = 0; t < 7; ++t) acc[t] = zero;

  for (int ks = 0; ks < 2; ++ks) {               // K = 256 in two halves
    for (int idx = tid; idx < 64 * 128; idx += 128) {
      int ol = idx >> 7, cl = idx & 127;
      int o = o0 + ol;
      const float* wp = (o < 128) ? (w_theta + (size_t)o * 256)
                      : (o < 256) ? (w_phi   + (size_t)(o - 128) * 256)
                                  : (w_g     + (size_t)(o - 256) * 256);
      w_lds[idx] = f2bf(wp[ks * 128 + cl]);
    }
    for (int idx = tid; idx < 112 * 128; idx += 128) {
      int cl = idx / 112, nl = idx % 112;
      x_lds[nl * 128 + cl] =
          f2bf(x[((size_t)b * 256 + ks * 128 + cl) * NPIX + n0 + nl]);
    }
    __syncthreads();
#pragma unroll
    for (int k = 0; k < 4; ++k) {
      v16bf a = load_frag_a(w_lds, 128, wave * 16, k * 32, lane);   // 2 DS reads
      v16bf bb[7];
#pragma unroll
      for (int t = 0; t < 7; ++t)
        bb[t] = load_frag_b(x_lds, 128, t * 16, k * 32, lane);      // 14 DS reads
#pragma unroll
      for (int t = 0; t < 7; ++t)
        acc[t] = wmma_bf16(a, bb[t], acc[t]);
    }
    sgb_pipeline<4, 16, 7>();   // DS16,DS16,MMA7,DS16,MMA7,DS16,MMA7,MMA7
    __syncthreads();
  }

  const int Mh = (lane >> 4) * 8, nc = lane & 15;
#pragma unroll
  for (int t = 0; t < 7; ++t) {
    int n = n0 + t * 16 + nc;
#pragma unroll
    for (int r = 0; r < 8; ++r) {
      int o = o0 + wave * 16 + r + Mh;
      int sec = o >> 7, och = o & 127;
      unsigned short v = f2bf(acc[t][r]);
      if (sec < 2) {  // theta / phi: (b, n, ch)
        tpg[(size_t)sec * SECN + ((size_t)b * NPIX + n) * 128 + och] = v;
      } else {        // g: (b, ch, n)
        tpg[2 * SECN + ((size_t)b * 128 + och) * NPIX + n] = v;
      }
    }
  }
}

// ---------------------------------------------------------------------------
// Kernel 2: fused exp-softmax attention, flash style.
// 64 query rows per WG, 49 chunks of 64 keys, double-buffered async LDS:
// per wave per chunk: 8 async b128 for phi + 8 for g (ASYNCcnt <= 32).
// LDS map (bytes): [0,16K) theta staging, aliased by per-wave exp(S) scratch;
//                  [16K,48K) phi x2; [48K,84K) g x2 ([144][64], ch128 = ones).
// ---------------------------------------------------------------------------
#define TH_USH 0
#define PH_USH 8192
#define G_USH  24576
#define PH_BUF 8192      // ushorts per phi buffer (64*128)
#define G_BUF  9216      // ushorts per g buffer (144*64)

__global__ __launch_bounds__(128) void attn_kernel(
    const unsigned short* __restrict__ theta_ws,
    const unsigned short* __restrict__ phi_ws,
    const unsigned short* __restrict__ g_ws,
    unsigned short* __restrict__ y_ws) {
  __shared__ __align__(16) unsigned short smem[43008];        // 84 KB
  const int tid = threadIdx.x, lane = tid & 31, wave = tid >> 5;
  const int n0 = blockIdx.x * 64;
  const int b  = blockIdx.y;
  const int Mh = (lane >> 4) * 8, nc = lane & 15;

  {  // theta tile once (16 KB, plain vectorized copy)
    const uint4* src = (const uint4*)(theta_ws + ((size_t)b * NPIX + n0) * 128);
    uint4* dst = (uint4*)smem;
    for (int i = tid; i < 1024; i += 128) dst[i] = src[i];
  }
  __syncthreads();

  v16bf a_th[4];  // theta A-fragments cached in registers for the whole sweep
#pragma unroll
  for (int k = 0; k < 4; ++k)
    a_th[k] = load_frag_a(smem + TH_USH, 128, wave * 16, k * 32, lane);

  // ones column (ch=128) and zero pad rows for both g buffers: static, fill once
  for (int idx = tid; idx < 2 * 16 * 64; idx += 128) {
    int bufi = idx >> 10, r = idx & 1023;
    int ch = 128 + (r >> 6), m = r & 63;
    smem[G_USH + bufi * G_BUF + ch * 64 + m] =
        (ch == 128) ? (unsigned short)0x3F80 : (unsigned short)0;
  }

  const unsigned smem_lds = (unsigned)(unsigned long long)(const void*)smem;
  const char* phi_plane = (const char*)(phi_ws + (size_t)b * NPIX * 128);
  const char* g_plane   = (const char*)(g_ws + (size_t)b * 128 * NPIX);
  const unsigned ph_off0 = (unsigned)(wave * 4096 + lane * 16);          // bytes
  const int ch0 = wave * 32 + (lane >> 3);
  const unsigned g_voff0 = (unsigned)(ch0 * (NPIX * 2) + (lane & 7) * 16);
  const unsigned g_lds0  = (unsigned)(ch0 * 128 + (lane & 7) * 16);

  // issue one chunk's copies for this wave: 8 phi + 8 g async b128 ops
  auto issue_chunk = [&](int m0, int bufi) {
    const void* pb = phi_plane + (size_t)m0 * 256;   // 128 bf16 per row
    unsigned plds = smem_lds + PH_USH * 2 + bufi * (PH_BUF * 2) + ph_off0;
#pragma unroll
    for (int i = 0; i < 8; ++i)
      async_copy16(plds + i * 512, ph_off0 + i * 512, pb);
    const void* gb = g_plane + (size_t)m0 * 2;       // ch-major rows, stride NPIX*2
    unsigned glds = smem_lds + G_USH * 2 + bufi * (G_BUF * 2) + g_lds0;
#pragma unroll
    for (int i = 0; i < 8; ++i)
      async_copy16(glds + i * 512, g_voff0 + i * (unsigned)(4 * NPIX * 2), gb);
  };

  v8f yacc[9];
  v8f zero = {};
#pragma unroll
  for (int t = 0; t < 9; ++t) yacc[t] = zero;

  issue_chunk(0, 0);                                  // prologue
  for (int c = 0; c < 49; ++c) {
    const int cur = c & 1;
    if (c < 48) {
      issue_chunk((c + 1) * 64, 1 - cur);             // prefetch next chunk
      wait_async_le16();                              // cur buffer retired
    } else {
      wait_async_0();
    }
    __syncthreads();                                  // cross-wave visibility

    const unsigned short* ph = smem + PH_USH + cur * PH_BUF;   // [m][c]
    const unsigned short* gg = smem + G_USH + cur * G_BUF;     // [ch][m]
    unsigned short* pw = smem + TH_USH + wave * 1024;          // [16][64] exp(S)

    // S = theta @ phi^T (16x64 per wave), batched 4-wide per k-step
    v8f s[4];
#pragma unroll
    for (int tc = 0; tc < 4; ++tc) s[tc] = zero;
#pragma unroll
    for (int k = 0; k < 4; ++k) {
      v16bf bb[4];
#pragma unroll
      for (int tc = 0; tc < 4; ++tc)
        bb[tc] = load_frag_b(ph, 128, tc * 16, k * 32, lane);   // 8 DS reads
#pragma unroll
      for (int tc = 0; tc < 4; ++tc)
        s[tc] = wmma_bf16(a_th[k], bb[tc], s[tc]);
    }
#pragma unroll
    for (int tc = 0; tc < 4; ++tc)
#pragma unroll
      for (int r = 0; r < 8; ++r)
        pw[(r + Mh) * 64 + tc * 16 + nc] = f2bf(__expf(s[tc][r]));

    // Y[16x144] += P[16x64] @ G[64x144]  (tile 8 column 0 = row sums)
#pragma unroll
    for (int kk = 0; kk < 2; ++kk) {
      v16bf pa = load_frag_a(pw, 64, 0, kk * 32, lane);         // 2 DS reads
#pragma unroll
      for (int tb = 0; tb < 3; ++tb) {               // batches of 3 tiles
        v16bf gb3[3];
#pragma unroll
        for (int u = 0; u < 3; ++u)
          gb3[u] = load_frag_b(gg, 64, (tb * 3 + u) * 16, kk * 32, lane);
#pragma unroll
        for (int u = 0; u < 3; ++u)
          yacc[tb * 3 + u] = wmma_bf16(pa, gb3[u], yacc[tb * 3 + u]);
      }
    }

    // Scheduling pipeline for this region (soft hints; deps still enforced):
    // S phase: DS8,DS8,MMA4,DS8,MMA4,DS8,MMA4,MMA4
    sgb_pipeline<4, 8, 4>();
    // exp(S) spill (32 ds_store_b16), VALU floats freely
    SGB(SGB_DSW, 32);
    // Y phase: DS20,DS20,MMA9,MMA9 (pa loads can't pass p stores: LDS aliasing)
    SGB(SGB_DSR, 20);
    SGB(SGB_DSR, 20);
    SGB(SGB_MMA, 9);
    SGB(SGB_MMA, 9);

    __syncthreads();   // all waves done reading before next chunk overwrites
  }

  // Normalize by row sums (yacc[8] N=0 lives in lanes 0/16) and store bf16 y.
#pragma unroll
  for (int t = 0; t < 8; ++t) {
#pragma unroll
    for (int r = 0; r < 8; ++r) {
      float l = __shfl(yacc[8][r], (lane >> 4) << 4, 32);
      float v = yacc[t][r] / l;
      int n = n0 + wave * 16 + r + Mh;
      y_ws[((size_t)b * NPIX + n) * 128 + t * 16 + nc] = f2bf(v);
    }
  }
}

// ---------------------------------------------------------------------------
// Kernel 3: z = W_z @ y + x (fp32 out).
// ---------------------------------------------------------------------------
__global__ __launch_bounds__(128) void zout_kernel(
    const unsigned short* __restrict__ y_ws, const float* __restrict__ w_z,
    const float* __restrict__ x, float* __restrict__ out) {
  __shared__ __align__(16) unsigned short w_lds[64 * 128];    // 16 KB [o][c]
  __shared__ __align__(16) unsigned short y_lds[112 * 128];   // 28 KB [n][c]
  const int tid = threadIdx.x, lane = tid & 31, wave = tid >> 5;
  const int n0 = blockIdx.x * 112;
  const int o0 = blockIdx.y * 64;
  const int b  = blockIdx.z;

  for (int idx = tid; idx < 64 * 128; idx += 128)
    w_lds[idx] = f2bf(w_z[(size_t)o0 * 128 + idx]);
  {
    const uint4* src = (const uint4*)(y_ws + ((size_t)b * NPIX + n0) * 128);
    uint4* dst = (uint4*)y_lds;
    for (int i = tid; i < 112 * 128 * 2 / 16; i += 128) dst[i] = src[i];
  }
  __syncthreads();

  v8f acc[7];
  v8f zero = {};
#pragma unroll
  for (int t = 0; t < 7; ++t) acc[t] = zero;
#pragma unroll
  for (int k = 0; k < 4; ++k) {
    v16bf a = load_frag_a(w_lds, 128, wave * 16, k * 32, lane);
    v16bf bb[7];
#pragma unroll
    for (int t = 0; t < 7; ++t)
      bb[t] = load_frag_b(y_lds, 128, t * 16, k * 32, lane);
#pragma unroll
    for (int t = 0; t < 7; ++t)
      acc[t] = wmma_bf16(a, bb[t], acc[t]);
  }
  sgb_pipeline<4, 16, 7>();

  const int Mh = (lane >> 4) * 8, nc = lane & 15;
#pragma unroll
  for (int t = 0; t < 7; ++t) {
    int n = n0 + t * 16 + nc;
#pragma unroll
    for (int r = 0; r < 8; ++r) {
      int o = o0 + wave * 16 + r + Mh;
      size_t off = ((size_t)b * 256 + o) * NPIX + n;
      out[off] = acc[t][r] + x[off];
    }
  }
}

extern "C" void kernel_launch(void* const* d_in, const int* in_sizes, int n_in,
                              void* d_out, int out_size, void* d_ws, size_t ws_size,
                              hipStream_t stream) {
  (void)in_sizes; (void)n_in; (void)out_size; (void)ws_size;
  const float* x       = (const float*)d_in[0];
  const float* w_theta = (const float*)d_in[1];
  const float* w_phi   = (const float*)d_in[2];
  const float* w_g     = (const float*)d_in[3];
  const float* w_z     = (const float*)d_in[4];
  float* out = (float*)d_out;

  unsigned short* ws = (unsigned short*)d_ws;
  unsigned short* tpg = ws;               // theta | phi | g planes (~19.3 MB)
  unsigned short* y   = ws + 3 * SECN;    // y bf16 (~6.4 MB)

  proj_kernel<<<dim3(28, 6, 8), 128, 0, stream>>>(x, w_theta, w_phi, w_g, tpg);
  attn_kernel<<<dim3(49, 8), 128, 0, stream>>>(tpg, tpg + SECN, tpg + 2 * SECN, y);
  zout_kernel<<<dim3(28, 4, 8), 128, 0, stream>>>(y, w_z, x, out);
}
